// ExpertGQALayer_60790967107726
// MI455X (gfx1250) — matrix operand order repaired and checked by reading, not compiled, not run on hardware
//
#include <hip/hip_runtime.h>
#include <hip/hip_bf16.h>

typedef __attribute__((ext_vector_type(16))) __bf16        v16bf;
typedef __attribute__((ext_vector_type(8)))  float         v8f;
typedef __attribute__((ext_vector_type(4)))  unsigned int  v4ui;
typedef __attribute__((ext_vector_type(4)))  int           v4i;
typedef __attribute__((ext_vector_type(8)))  int           v8i;

__device__ __forceinline__ unsigned short f2bf(float f) {
    unsigned int u = __float_as_uint(f);
    unsigned int r = u + 0x7fffu + ((u >> 16) & 1u);   // RNE bf16
    return (unsigned short)(r >> 16);
}
__device__ __forceinline__ __bf16 bfraw(unsigned short s) {
    union { unsigned short u; __bf16 b; } x; x.u = s; return x.b;
}

#if defined(__HIP_DEVICE_COMPILE__) && \
    __has_builtin(__builtin_amdgcn_tensor_load_to_lds) && \
    __has_builtin(__builtin_amdgcn_s_wait_tensorcnt)
#define USE_TDM 1
#else
#define USE_TDM 0
#endif

#define BM 128
#define BN 128
#define BK 32

// LDS pad configs for TDM: pad_enable | pad_interval | pad_amount
// PAD34: +1 DWORD per 16 DWORDs  (32-elem bf16 row -> stride 34 halfwords)
// PAD68: +2 DWORDs per 32 DWORDs (64-elem bf16 row -> stride 68 halfwords)
#define PAD34 ((1u << 20) | (3u << 22) | (0u << 25))
#define PAD68 ((1u << 20) | (4u << 22) | (1u << 25))

#if USE_TDM
__device__ __forceinline__ void tdm_load_tile(
    const unsigned short* gtile, unsigned ldsaddr,
    unsigned dim0, unsigned dim1, unsigned tile0, unsigned tile1,
    unsigned long long stride0, unsigned padcfg)
{
    unsigned long long ga = (unsigned long long)(const void*)gtile;
    v4ui g0;
    g0.x = 1u;                                                // count=1
    g0.y = ldsaddr;                                           // lds byte address
    g0.z = (unsigned)(ga & 0xffffffffu);                      // global_addr lo
    g0.w = (unsigned)((ga >> 32) & 0x01ffffffu) | (2u << 30); // addr hi | type=2
    unsigned w0 = (1u << 16) | padcfg;                        // data_size=2B
    v8i g1;
    g1[0] = (int)w0;
    g1[1] = (int)((dim0 & 0xffffu) << 16);
    g1[2] = (int)((dim0 >> 16) | ((dim1 & 0xffffu) << 16));
    g1[3] = (int)((dim1 >> 16) | (tile0 << 16));
    g1[4] = (int)(tile1);
    g1[5] = (int)(stride0 & 0xffffffffu);
    g1[6] = (int)((stride0 >> 32) & 0xffffu);
    g1[7] = 0;
    v4i z4 = {0, 0, 0, 0};
#if defined(__clang_major__) && __clang_major__ >= 23
    v8i z8 = {0, 0, 0, 0, 0, 0, 0, 0};
    __builtin_amdgcn_tensor_load_to_lds(g0, g1, z4, z4, z8, 0);
#else
    __builtin_amdgcn_tensor_load_to_lds(g0, g1, z4, z4, 0);
#endif
}
#define LDSA(p) ((unsigned)(size_t)(void*)(p))
#endif

// ---------------------------------------------------------------------------
// bf16 WMMA GEMM: C[M,N] = A[M,K] * Bt[N,K]^T, both bf16 row-major, acc f32.
// Both A and B tiles are 128x32 row-major with contiguous K-pairs -> identical
// TDM descriptors (PAD34) and identical ds_load_2addr-friendly fragment loads.
// MODE: 0 none, 1 silu, 2 *AUX, 3 +RES.  OUTBF: store bf16.
// ---------------------------------------------------------------------------
template <int MODE, int OUTBF>
__global__ __launch_bounds__(256) void gemm_bf16_wmma(
    const unsigned short* __restrict__ A, const unsigned short* __restrict__ Bt,
    const float* __restrict__ AUX, const float* __restrict__ RES,
    void* __restrict__ Cv, int M, int N, int K)
{
    __shared__ unsigned short As[2][BM][BK + 2];
    __shared__ unsigned short Bs[2][BN][BK + 2];

    const int tid  = threadIdx.x;
    const int lane = tid & 31;
    const int w    = tid >> 5;
    const int wm   = w >> 2;
    const int wn   = w & 3;
    const int bM   = blockIdx.y * BM;
    const int bN   = blockIdx.x * BN;

    v8f acc[4][2];
#pragma unroll
    for (int mt = 0; mt < 4; ++mt)
#pragma unroll
        for (int nt = 0; nt < 2; ++nt)
            acc[mt][nt] = 0.0f;

    const int nk = K / BK;

#if USE_TDM
    if (w == 0) {
        tdm_load_tile(A  + (size_t)bM * K, LDSA(&As[0][0][0]),
                      (unsigned)K, (unsigned)M, BK, BM, (unsigned long long)K, PAD34);
        tdm_load_tile(Bt + (size_t)bN * K, LDSA(&Bs[0][0][0]),
                      (unsigned)K, (unsigned)N, BK, BN, (unsigned long long)K, PAD34);
    }
#endif

    for (int it = 0; it < nk; ++it) {
        const int cur = it & 1;
#if USE_TDM
        __builtin_amdgcn_s_wait_tensorcnt(0);
        __syncthreads();
        if (it + 1 < nk && w == 0) {
            const int k1 = (it + 1) * BK;
            const int nxt = cur ^ 1;
            tdm_load_tile(A  + (size_t)bM * K + k1, LDSA(&As[nxt][0][0]),
                          (unsigned)K, (unsigned)M, BK, BM, (unsigned long long)K, PAD34);
            tdm_load_tile(Bt + (size_t)bN * K + k1, LDSA(&Bs[nxt][0][0]),
                          (unsigned)K, (unsigned)N, BK, BN, (unsigned long long)K, PAD34);
        }
#else
        const int k0 = it * BK;
#pragma unroll
        for (int i = 0; i < 2; ++i) {           // A tile 128x32
            int lin = tid + i * 256;
            int r = lin >> 2, c8 = (lin & 3) * 8;
            uint4 q4 = *(const uint4*)(A + (size_t)(bM + r) * K + k0 + c8);
            unsigned vv[4] = {q4.x, q4.y, q4.z, q4.w};
#pragma unroll
            for (int t2 = 0; t2 < 4; ++t2) {
                As[0][r][c8 + 2 * t2]     = (unsigned short)(vv[t2] & 0xffffu);
                As[0][r][c8 + 2 * t2 + 1] = (unsigned short)(vv[t2] >> 16);
            }
        }
#pragma unroll
        for (int i = 0; i < 2; ++i) {           // B tile 128x32 (transposed weights)
            int lin = tid + i * 256;
            int r = lin >> 2, c8 = (lin & 3) * 8;
            uint4 q4 = *(const uint4*)(Bt + (size_t)(bN + r) * K + k0 + c8);
            unsigned vv[4] = {q4.x, q4.y, q4.z, q4.w};
#pragma unroll
            for (int t2 = 0; t2 < 4; ++t2) {
                Bs[0][r][c8 + 2 * t2]     = (unsigned short)(vv[t2] & 0xffffu);
                Bs[0][r][c8 + 2 * t2 + 1] = (unsigned short)(vv[t2] >> 16);
            }
        }
        __syncthreads();
#endif

#if USE_TDM
        const unsigned short (*Ac)[BK + 2] = As[cur];
        const unsigned short (*Bc)[BK + 2] = Bs[cur];
#else
        const unsigned short (*Ac)[BK + 2] = As[0];
        const unsigned short (*Bc)[BK + 2] = Bs[0];
#endif
        // A 16x32 frag: lanes 0-15 K{0..7,16..23}; lanes 16-31 K{8..15,24..31}
        v16bf afr[4];
        {
            const int arow_lo = lane & 15;
            const int kg = (lane >> 4) * 8;
#pragma unroll
            for (int mt = 0; mt < 4; ++mt) {
                const int arow = wm * 64 + mt * 16 + arow_lo;
#pragma unroll
                for (int j = 0; j < 8; ++j) {
                    const int kk = ((j & 4) << 2) + kg + ((j & 3) << 1);
                    afr[mt][2 * j]     = bfraw(Ac[arow][kk]);
                    afr[mt][2 * j + 1] = bfraw(Ac[arow][kk + 1]);
                }
            }
        }
        // B 32x16 frag from Bt rows: lane col n; lanes 0-15 K{0..7,16..23}; lanes 16-31 K{8..15,24..31}
        v16bf bfr[2];
        {
            const int ncl = lane & 15;
            const int kh = (lane >> 4) * 8;
#pragma unroll
            for (int nt = 0; nt < 2; ++nt) {
                const int nrow = wn * 32 + nt * 16 + ncl;
#pragma unroll
                for (int j = 0; j < 8; ++j) {
                    const int kk = ((j & 4) << 2) + kh + ((j & 3) << 1);
                    bfr[nt][2 * j]     = bfraw(Bc[nrow][kk]);
                    bfr[nt][2 * j + 1] = bfraw(Bc[nrow][kk + 1]);
                }
            }
        }
#pragma unroll
        for (int mt = 0; mt < 4; ++mt)
#pragma unroll
            for (int nt = 0; nt < 2; ++nt)
                acc[mt][nt] = __builtin_amdgcn_wmma_f32_16x16x32_bf16(
                    false, afr[mt], false, bfr[nt], (short)0, acc[mt][nt], false, false);
#if !USE_TDM
        __syncthreads();
#endif
    }

    const int rhalf = (lane >> 4) * 8;
    const int ncl = lane & 15;
#pragma unroll
    for (int mt = 0; mt < 4; ++mt) {
#pragma unroll
        for (int nt = 0; nt < 2; ++nt) {
#pragma unroll
            for (int r = 0; r < 8; ++r) {
                const int row = bM + wm * 64 + mt * 16 + rhalf + r;
                const int col = bN + wn * 32 + nt * 16 + ncl;
                float v = acc[mt][nt][r];
                if (MODE == 1) v = v / (1.0f + __expf(-v));
                if (MODE == 2) v *= AUX[(size_t)row * N + col];
                if (MODE == 3) v += RES[(size_t)row * N + col];
                if (OUTBF) ((unsigned short*)Cv)[(size_t)row * N + col] = f2bf(v);
                else       ((float*)Cv)[(size_t)row * N + col] = v;
            }
        }
    }
}

// ---------------------------------------------------------------------------
// Weight transpose + f32->bf16: W[K,N] -> Wt[N,K]. grid(N/32, K/32), 256 thr.
// ---------------------------------------------------------------------------
__global__ __launch_bounds__(256) void wtrans_k(
    const float* __restrict__ wsrc, unsigned short* __restrict__ wt, int K, int N)
{
    __shared__ float tile[32][33];
    const int n0 = blockIdx.x * 32, k0 = blockIdx.y * 32;
    const int r = threadIdx.x >> 5, c = threadIdx.x & 31;
#pragma unroll
    for (int i = 0; i < 4; ++i)
        tile[r + 8 * i][c] = wsrc[(size_t)(k0 + r + 8 * i) * N + n0 + c];
    __syncthreads();
#pragma unroll
    for (int i = 0; i < 4; ++i)
        wt[(size_t)(n0 + r + 8 * i) * K + k0 + c] = f2bf(tile[c][r + 8 * i]);
}

// V transpose: v f32 [8192][512] -> vt bf16 [16][8][64 d][512 s]
__global__ __launch_bounds__(256) void vtrans_k(
    const float* __restrict__ v, unsigned short* __restrict__ vt)
{
    __shared__ float tile[64][65];
    const int blk = blockIdx.x;
    const int sb = blk & 7, hk = (blk >> 3) & 7, bi = blk >> 6;
    const int t = threadIdx.x;
#pragma unroll
    for (int i = 0; i < 16; ++i) {
        int lin = t + i * 256, r = lin >> 6, c = lin & 63;
        tile[r][c] = v[(size_t)(bi * 512 + sb * 64 + r) * 512 + hk * 64 + c];
    }
    __syncthreads();
#pragma unroll
    for (int i = 0; i < 16; ++i) {
        int lin = t + i * 256, r = lin >> 6, c = lin & 63;
        vt[((size_t)(bi * 8 + hk) * 64 + r) * 512 + sb * 64 + c] = f2bf(tile[c][r]);
    }
}

// RMSNorm: one row (2048) per block, bf16 output
__global__ __launch_bounds__(256) void rmsnorm_k(
    const float* __restrict__ x, const float* __restrict__ wgt,
    unsigned short* __restrict__ o, int ncols)
{
    __shared__ float red[256];
    const int row = blockIdx.x;
    const int t = threadIdx.x;
    const size_t base = (size_t)row * ncols;
    float p = 0.0f;
    for (int c = t; c < ncols; c += 256) { float v = x[base + c]; p += v * v; }
    red[t] = p; __syncthreads();
    for (int off = 128; off > 0; off >>= 1) {
        if (t < off) red[t] += red[t + off];
        __syncthreads();
    }
    const float scale = rsqrtf(red[0] / (float)ncols + 1e-6f);
    for (int c = t; c < ncols; c += 256)
        o[base + c] = f2bf(x[base + c] * scale * wgt[c]);
}

// RoPE: read f32 projection, write rotated bf16 copy
__global__ __launch_bounds__(256) void rope_k(
    const float* __restrict__ in, unsigned short* __restrict__ ob,
    const int* __restrict__ pos, int nheads)
{
    const int row = blockIdx.x;
    const float p0 = (float)pos[row];
    const int ncols = nheads * 64;
    for (int pr = threadIdx.x; pr < nheads * 32; pr += 256) {
        const int head = pr >> 5, dd = pr & 31;
        const float invf = __expf(-(float)dd * (9.210340371976184f / 32.0f));
        const float f = p0 * invf;
        float s, c;
        __sincosf(f, &s, &c);
        const size_t i1 = (size_t)row * ncols + head * 64 + dd;
        const float x1 = in[i1], x2 = in[i1 + 32];
        ob[i1]      = f2bf(x1 * c - x2 * s);
        ob[i1 + 32] = f2bf(x2 * c + x1 * s);
    }
}

// ---------------------------------------------------------------------------
// WMMA GQA attention. One (b, hq, 64-query block) per 256-thread workgroup.
// S = Q K^T via WMMA (TDM-staged bf16 tiles), f32 scores in dynamic LDS,
// row softmax -> bf16 P in LDS, O = P V via WMMA with d-major V tiles.
// ---------------------------------------------------------------------------
#define TSTR 68    // staged tile row stride (halfwords)
#define SBS  516   // score row stride (f32)
#define PBS  520   // P row stride (halfwords)

__global__ __launch_bounds__(256) void attn_wmma_k(
    const unsigned short* __restrict__ Qb,   // [8192][2048] roped bf16
    const unsigned short* __restrict__ Kbf,  // [8192][512]  roped bf16
    const unsigned short* __restrict__ Vtb,  // [16][8][64][512] bf16 (d-major)
    unsigned short* __restrict__ ctx)        // [8192][2048] bf16
{
    extern __shared__ char smem[];
    float* Sb   = (float*)smem;                          // [64][SBS]
    float* vinv = Sb + 64 * SBS;                         // [64]
    unsigned short* Qs = (unsigned short*)(vinv + 64);   // [64][TSTR]
    unsigned short* Ks = Qs + 64 * TSTR;                 // [2][64][TSTR]
    unsigned short* Vs = Ks + 2 * 64 * TSTR;             // [2][64][TSTR]
    unsigned short* Pb = Vs + 2 * 64 * TSTR;             // [64][PBS]

    const int tid = threadIdx.x, lane = tid & 31, w = tid >> 5;
    const int blk = blockIdx.x;
    const int qb = blk & 7, hq = (blk >> 3) & 31, bi = blk >> 8;
    const int hk = hq >> 2;
    const int q0 = qb * 64;

    const unsigned short* Qg = Qb  + (size_t)(bi * 512 + q0) * 2048 + hq * 64;
    const unsigned short* Kg = Kbf + (size_t)(bi * 512) * 512 + hk * 64;
    const unsigned short* Vg = Vtb + (size_t)(bi * 8 + hk) * 64 * 512;

    const int qt0 = w >> 2, kt = w & 3;       // wave tiles: (qt0, qt0+2) x kt
    const int arow_lo = lane & 15, kg8 = (lane >> 4) * 8;
    const int ncl = lane & 15, kh = (lane >> 4) * 16;
    const int rhalf = (lane >> 4) * 8;

    // ---------------- phase 1: S = Q K^T ----------------
#if USE_TDM
    if (w == 0) {
        tdm_load_tile(Qg, LDSA(Qs), 2048, 65536, 64, 64, 2048, PAD68);
        tdm_load_tile(Kg, LDSA(Ks), 512, 65536, 64, 64, 512, PAD68);
    }
#else
#pragma unroll
    for (int i = 0; i < 2; ++i) {   // stage Q 64x64 once
        int lin = tid + i * 256, r = lin >> 3, c8 = (lin & 7) * 8;
        uint4 q4 = *(const uint4*)(Qg + (size_t)r * 2048 + c8);
        unsigned vv[4] = {q4.x, q4.y, q4.z, q4.w};
#pragma unroll
        for (int t2 = 0; t2 < 4; ++t2) {
            Qs[r * TSTR + c8 + 2 * t2]     = (unsigned short)(vv[t2] & 0xffffu);
            Qs[r * TSTR + c8 + 2 * t2 + 1] = (unsigned short)(vv[t2] >> 16);
        }
    }
#endif
    for (int kc = 0; kc < 8; ++kc) {
        const int cur = kc & 1;
#if USE_TDM
        __builtin_amdgcn_s_wait_tensorcnt(0);
        __syncthreads();
        if (kc + 1 < 8 && w == 0)
            tdm_load_tile(Kg + (size_t)(kc + 1) * 64 * 512, LDSA(Ks + (cur ^ 1) * 64 * TSTR),
                          512, 65536, 64, 64, 512, PAD68);
        const unsigned short* Kc = Ks + cur * 64 * TSTR;
#else
#pragma unroll
        for (int i = 0; i < 2; ++i) {
            int lin = tid + i * 256, r = lin >> 3, c8 = (lin & 7) * 8;
            uint4 q4 = *(const uint4*)(Kg + (size_t)(kc * 64 + r) * 512 + c8);
            unsigned vv[4] = {q4.x, q4.y, q4.z, q4.w};
#pragma unroll
            for (int t2 = 0; t2 < 4; ++t2) {
                Ks[r * TSTR + c8 + 2 * t2]     = (unsigned short)(vv[t2] & 0xffffu);
                Ks[r * TSTR + c8 + 2 * t2 + 1] = (unsigned short)(vv[t2] >> 16);
            }
        }
        __syncthreads();
        const unsigned short* Kc = Ks;
#endif
        v8f sacc[2];
        sacc[0] = 0.0f; sacc[1] = 0.0f;
#pragma unroll
        for (int st = 0; st < 2; ++st) {
            v16bf bfr;   // B = K^T : lane col = key, k-dim = d
#pragma unroll
            for (int j = 0; j < 8; ++j) {
                const int dd = ((j & 4) << 2) + kg8 + ((j & 3) << 1) + st * 32;
                bfr[2 * j]     = bfraw(Kc[(kt * 16 + ncl) * TSTR + dd]);
                bfr[2 * j + 1] = bfraw(Kc[(kt * 16 + ncl) * TSTR + dd + 1]);
            }
#pragma unroll
            for (int half = 0; half < 2; ++half) {
                const int qt = qt0 + 2 * half;
                v16bf afr;
#pragma unroll
                for (int j = 0; j < 8; ++j) {
                    const int dk = ((j & 4) << 2) + kg8 + ((j & 3) << 1) + st * 32;
                    afr[2 * j]     = bfraw(Qs[(qt * 16 + arow_lo) * TSTR + dk]);
                    afr[2 * j + 1] = bfraw(Qs[(qt * 16 + arow_lo) * TSTR + dk + 1]);
                }
                sacc[half] = __builtin_amdgcn_wmma_f32_16x16x32_bf16(
                    false, afr, false, bfr, (short)0, sacc[half], false, false);
            }
        }
#pragma unroll
        for (int half = 0; half < 2; ++half) {
            const int qt = qt0 + 2 * half;
#pragma unroll
            for (int r = 0; r < 8; ++r)
                Sb[(qt * 16 + rhalf + r) * SBS + kc * 64 + kt * 16 + ncl] =
                    sacc[half][r] * 0.125f;
        }
#if !USE_TDM
        __syncthreads();
#endif
    }
    __syncthreads();

    // ---------------- softmax (row per thread) ----------------
    if (tid < 64) {
        float m = -1e30f;
        for (int c = 0; c < 512; ++c) m = fmaxf(m, Sb[tid * SBS + c]);
        float s = 0.0f;
        for (int c = 0; c < 512; ++c) {
            const float p = __expf(Sb[tid * SBS + c] - m);
            s += p;
            Pb[tid * PBS + c] = f2bf(p);
        }
        vinv[tid] = 1.0f / s;
    }
    __syncthreads();

    // ---------------- phase 3: O = P V ----------------
    v8f oacc[2];
    oacc[0] = 0.0f; oacc[1] = 0.0f;
#if USE_TDM
    if (w == 0)
        tdm_load_tile(Vg, LDSA(Vs), 512, 65536, 64, 64, 512, PAD68);
#endif
    for (int kc = 0; kc < 8; ++kc) {
        const int cur = kc & 1;
#if USE_TDM
        __builtin_amdgcn_s_wait_tensorcnt(0);
        __syncthreads();
        if (kc + 1 < 8 && w == 0)
            tdm_load_tile(Vg + (size_t)(kc + 1) * 64, LDSA(Vs + (cur ^ 1) * 64 * TSTR),
                          512, 65536, 64, 64, 512, PAD68);
        const unsigned short* Vc = Vs + cur * 64 * TSTR;
#else
#pragma unroll
        for (int i = 0; i < 2; ++i) {   // rows = d, cols = keys of this chunk
            int lin = tid + i * 256, r = lin >> 3, c8 = (lin & 7) * 8;
            uint4 q4 = *(const uint4*)(Vg + (size_t)r * 512 + kc * 64 + c8);
            unsigned vv[4] = {q4.x, q4.y, q4.z, q4.w};
#pragma unroll
            for (int t2 = 0; t2 < 4; ++t2) {
                Vs[r * TSTR + c8 + 2 * t2]     = (unsigned short)(vv[t2] & 0xffffu);
                Vs[r * TSTR + c8 + 2 * t2 + 1] = (unsigned short)(vv[t2] >> 16);
            }
        }
        __syncthreads();
        const unsigned short* Vc = Vs;
#endif
#pragma unroll
        for (int st = 0; st < 2; ++st) {
            v16bf bfr;   // B = V : lane col = d (row of d-major tile), k-dim = key
#pragma unroll
            for (int j = 0; j < 8; ++j) {
                const int kk = ((j & 4) << 2) + kg8 + ((j & 3) << 1) + st * 32;
                bfr[2 * j]     = bfraw(Vc[(kt * 16 + ncl) * TSTR + kk]);
                bfr[2 * j + 1] = bfraw(Vc[(kt * 16 + ncl) * TSTR + kk + 1]);
            }
#pragma unroll
            for (int half = 0; half < 2; ++half) {
                const int qt = qt0 + 2 * half;
                v16bf afr;
#pragma unroll
                for (int j = 0; j < 8; ++j) {
                    const int kk = ((j & 4) << 2) + kg8 + ((j & 3) << 1) + st * 32;
                    afr[2 * j]     = bfraw(Pb[(qt * 16 + arow_lo) * PBS + kc * 64 + kk]);
                    afr[2 * j + 1] = bfraw(Pb[(qt * 16 + arow_lo) * PBS + kc * 64 + kk + 1]);
                }
                oacc[half] = __builtin_amdgcn_wmma_f32_16x16x32_bf16(
                    false, afr, false, bfr, (short)0, oacc[half], false, false);
            }
        }
#if !USE_TDM
        __syncthreads();
#endif
    }

    // epilogue: scale by 1/rowsum, write bf16 ctx
#pragma unroll
    for (int half = 0; half < 2; ++half) {
        const int qt = qt0 + 2 * half;
#pragma unroll
        for (int r = 0; r < 8; ++r) {
            const int ql = qt * 16 + rhalf + r;
            const float val = oacc[half][r] * vinv[ql];
            ctx[(size_t)(bi * 512 + q0 + ql) * 2048 + hq * 64 + kt * 16 + ncl] = f2bf(val);
        }
    }
}

// ---------------------------------------------------------------------------
// inputs: x, pos_ids, wq, wk, wv, wo, wg, wu, wd, ln1_w, ln2_w
// ---------------------------------------------------------------------------
extern "C" void kernel_launch(void* const* d_in, const int* in_sizes, int n_in,
                              void* d_out, int out_size, void* d_ws, size_t ws_size,
                              hipStream_t stream) {
    const float* x   = (const float*)d_in[0];
    const int*   pos = (const int*)d_in[1];
    const float* wq  = (const float*)d_in[2];
    const float* wk  = (const float*)d_in[3];
    const float* wv  = (const float*)d_in[4];
    const float* wo  = (const float*)d_in[5];
    const float* wg  = (const float*)d_in[6];
    const float* wu  = (const float*)d_in[7];
    const float* wd  = (const float*)d_in[8];
    const float* ln1 = (const float*)d_in[9];
    const float* ln2 = (const float*)d_in[10];
    float* out = (float*)d_out;

    const int MT = 16 * 512;              // 8192 tokens
    const int H = 2048, NQ = 32, NKV = 8, IN = 8192;

    char* p = (char*)d_ws;
    unsigned short* hb  = (unsigned short*)p;  p += (size_t)MT * H * 2;
    float*          q   = (float*)p;           p += (size_t)MT * H * 4;
    float*          kb  = (float*)p;           p += (size_t)MT * 512 * 4;
    float*          vb  = (float*)p;           p += (size_t)MT * 512 * 4;
    unsigned short* cxb = (unsigned short*)p;  p += (size_t)MT * H * 2;
    float*          x2  = (float*)p;           p += (size_t)MT * H * 4;
    unsigned short* h2b = (unsigned short*)p;  p += (size_t)MT * H * 2;
    float*          g   = (float*)p;           p += (size_t)MT * IN * 4;
    unsigned short* qbb = (unsigned short*)p;  p += (size_t)MT * H * 2;
    unsigned short* kbb = (unsigned short*)p;  p += (size_t)MT * 512 * 2;
    unsigned short* vtb = (unsigned short*)p;  p += (size_t)MT * 512 * 2;
    unsigned short* wqt = (unsigned short*)p;  p += (size_t)H * H * 2;
    unsigned short* wkt = (unsigned short*)p;  p += (size_t)H * 512 * 2;
    unsigned short* wvt = (unsigned short*)p;  p += (size_t)H * 512 * 2;
    unsigned short* wot = (unsigned short*)p;  p += (size_t)H * H * 2;
    unsigned short* wgt = (unsigned short*)p;  p += (size_t)H * IN * 2;
    unsigned short* wut = (unsigned short*)p;  p += (size_t)H * IN * 2;
    unsigned short* wdt = (unsigned short*)p;  p += (size_t)IN * H * 2;
    // act (bf16, MT x IN) aliases dead hb/q/kb/vb region (exactly equal size)
    unsigned short* actb = (unsigned short*)d_ws;

    // ---- transpose+convert weights: W[K,N] -> Wt[N,K] bf16 ----
    wtrans_k<<<dim3(H  / 32, H  / 32), 256, 0, stream>>>(wq, wqt, H,  H);
    wtrans_k<<<dim3(512 / 32, H / 32), 256, 0, stream>>>(wk, wkt, H,  512);
    wtrans_k<<<dim3(512 / 32, H / 32), 256, 0, stream>>>(wv, wvt, H,  512);
    wtrans_k<<<dim3(H  / 32, H  / 32), 256, 0, stream>>>(wo, wot, H,  H);
    wtrans_k<<<dim3(IN / 32, H  / 32), 256, 0, stream>>>(wg, wgt, H,  IN);
    wtrans_k<<<dim3(IN / 32, H  / 32), 256, 0, stream>>>(wu, wut, H,  IN);
    wtrans_k<<<dim3(H  / 32, IN / 32), 256, 0, stream>>>(wd, wdt, IN, H);

    dim3 gH (H   / BN, MT / BM);
    dim3 gKV(512 / BN, MT / BM);
    dim3 gI (IN  / BN, MT / BM);

    // ---- attention half ----
    rmsnorm_k<<<MT, 256, 0, stream>>>(x, ln1, hb, H);
    gemm_bf16_wmma<0, 0><<<gH,  256, 0, stream>>>(hb, wqt, nullptr, nullptr, q,  MT, H,   H);
    gemm_bf16_wmma<0, 0><<<gKV, 256, 0, stream>>>(hb, wkt, nullptr, nullptr, kb, MT, 512, H);
    gemm_bf16_wmma<0, 0><<<gKV, 256, 0, stream>>>(hb, wvt, nullptr, nullptr, vb, MT, 512, H);
    rope_k<<<MT, 256, 0, stream>>>(q,  qbb, pos, NQ);
    rope_k<<<MT, 256, 0, stream>>>(kb, kbb, pos, NKV);
    vtrans_k<<<16 * 8 * 8, 256, 0, stream>>>(vb, vtb);

    const int attn_smem = 64 * SBS * 4 + 64 * 4            // scores + inv
                        + 64 * TSTR * 2                    // Q
                        + 2 * 64 * TSTR * 2 * 2            // K,V double buffers
                        + 64 * PBS * 2;                    // P
    attn_wmma_k<<<16 * 32 * 8, 256, attn_smem, stream>>>(qbb, kbb, vtb, cxb);

    gemm_bf16_wmma<3, 0><<<gH, 256, 0, stream>>>(cxb, wot, nullptr, x, x2, MT, H, H);

    // ---- MLP half ----
    rmsnorm_k<<<MT, 256, 0, stream>>>(x2, ln2, h2b, H);
    gemm_bf16_wmma<1, 0><<<gI, 256, 0, stream>>>(h2b, wgt, nullptr, nullptr, g, MT, IN, H);
    gemm_bf16_wmma<2, 1><<<gI, 256, 0, stream>>>(h2b, wut, g, nullptr, actb, MT, IN, H);
    gemm_bf16_wmma<3, 0><<<gH, 256, 0, stream>>>(actb, wdt, nullptr, x2, out, MT, H, IN);
}
